// NAO_14637248545385
// MI455X (gfx1250) — matrix operand order repaired and checked by reading, not compiled, not run on hardware
//
#include <hip/hip_runtime.h>
#include <hip/hip_bf16.h>
#include <math.h>

typedef __attribute__((ext_vector_type(16))) _Float16 v16h;
typedef __attribute__((ext_vector_type(8)))  _Float16 v8h;
typedef __attribute__((ext_vector_type(8)))  float    v8f;

#define DEV static __device__ __forceinline__

// ---------------- WMMA fragment loaders (CDNA5 16-bit layouts, ISA 7.12.2) ----

// A fragment (16x32 f16) from row-major A (lda elems/row), tile at (m0,k0).
// Lane layout: m = lane&15; grp = lane>>4.
//   a[2v+h], v<4 : K = grp*8 + 2v+h          -> 8 contiguous halves at k0+grp*8
//   a[2v+h], v>=4: K = 16 + grp*8 + 2(v-4)+h -> 8 contiguous halves at k0+16+grp*8
DEV v16h load_a_frag(const _Float16* __restrict__ A, int lda, int m0, int k0, int lane) {
  int m = lane & 15;
  int grp = (lane >> 4) & 1;
  const _Float16* p = A + (size_t)(m0 + m) * lda + k0 + grp * 8;
  v8h lo = *(const v8h*)p;
  v8h hi = *(const v8h*)(p + 16);
  v16h a;
#pragma unroll
  for (int i = 0; i < 8; ++i) { a[i] = lo[i]; a[8 + i] = hi[i]; }
  return a;
}

// B fragment (32x16 f16) read from BT stored N x K row-major (ldb = K stride).
// Lane layout: n = lane&15; grp = lane>>4; b[i] = BT[n][grp*8+i], b[8+i] = BT[n][16+grp*8+i].
DEV v16h load_bt_frag(const _Float16* __restrict__ BT, int ldb, int n0, int k0, int lane) {
  int n = lane & 15;
  int grp = (lane >> 4) & 1;
  const _Float16* p = BT + (size_t)(n0 + n) * ldb + k0 + grp * 8;
  v8h lo = *(const v8h*)p;
  v8h hi = *(const v8h*)(p + 16);
  v16h b;
#pragma unroll
  for (int i = 0; i < 8; ++i) { b[i] = lo[i]; b[8 + i] = hi[i]; }
  return b;
}

DEV v8f wmma_f32(v16h a, v16h b, v8f c) {
  return __builtin_amdgcn_wmma_f32_16x16x32_f16(false, a, false, b, (short)0, c, false, false);
}

// ---------------- GEMM: C[M,N] = alpha * A[M,K] * B[K,N] (+bias), BT = N x K ----
// One wave (32 threads) computes a 16x64 strip (A frag reused across 4 B tiles).
// blockIdx.z selects the batch (strides may be 0 for shared operands).
// If Ct != nullptr (requires gridDim.z==1), the result is ALSO stored transposed
// per 4096-row activation batch: Ct[bat][col][row] with contiguous 16B stores.
template <bool OUT_F16>
__global__ void gemm_bt_kernel(const _Float16* __restrict__ A, int lda, size_t a_bstride,
                               const _Float16* __restrict__ BT, int ldb, size_t bt_bstride,
                               void* __restrict__ Cv, int ldc, size_t c_bstride,
                               const float* __restrict__ bias, float alpha, int K,
                               _Float16* __restrict__ Ct, size_t ct_bstride) {
  int lane = threadIdx.x & 31;
  int m0 = blockIdx.x * 16;
  int n0 = blockIdx.y * 64;
  int bat = blockIdx.z;
  const _Float16* Ab = A + (size_t)bat * a_bstride;
  const _Float16* Bb = BT + (size_t)bat * bt_bstride;
  v8f acc[4] = {};
  for (int k0 = 0; k0 < K; k0 += 32) {
    v16h a = load_a_frag(Ab, lda, m0, k0, lane);
#pragma unroll
    for (int t = 0; t < 4; ++t) {
      v16h b = load_bt_frag(Bb, ldb, n0 + 16 * t, k0, lane);
      acc[t] = wmma_f32(a, b, acc[t]);
    }
  }
  int n = lane & 15, mg = (lane >> 4) * 8;
#pragma unroll
  for (int t = 0; t < 4; ++t)
#pragma unroll
    for (int v = 0; v < 8; ++v) {
      int col = n0 + 16 * t + n;
      float val = acc[t][v] * alpha;
      if (bias) val += bias[col];
      size_t idx = (size_t)bat * c_bstride + (size_t)(m0 + mg + v) * ldc + col;
      if (OUT_F16) ((_Float16*)Cv)[idx] = (_Float16)val;
      else         ((float*)Cv)[idx] = val;
    }
  if (Ct) {
    int bt2 = m0 >> 12;       // 4096 rows per activation batch
    int rloc = (m0 & 4095) + mg;
#pragma unroll
    for (int t = 0; t < 4; ++t) {
      v8h p;
#pragma unroll
      for (int v = 0; v < 8; ++v) {
        float val = acc[t][v] * alpha;
        if (bias) val += bias[n0 + 16 * t + n];
        p[v] = (_Float16)val;
      }
      *(v8h*)(Ct + (size_t)bt2 * ct_bstride + (size_t)(n0 + 16 * t + n) * 4096 + rloc) = p;
    }
  }
}

// ---------------- Q/K projection: Q row-major, K stored transposed ------------
// grid (R/16, 2): y==0 -> Q half (cols 0..63 of BT), y==1 -> K half, stored as
// KT[bat][k][row] (B x 64 x 4096) via contiguous v8h stores.
__global__ void gemm_qkt_kernel(const _Float16* __restrict__ A, int lda,
                                const _Float16* __restrict__ BT, int ldb,
                                _Float16* __restrict__ Qd, _Float16* __restrict__ KTd,
                                int K) {
  int lane = threadIdx.x & 31;
  int m0 = blockIdx.x * 16;
  int n0 = blockIdx.y * 64;
  v8f acc[4] = {};
  for (int k0 = 0; k0 < K; k0 += 32) {
    v16h a = load_a_frag(A, lda, m0, k0, lane);
#pragma unroll
    for (int t = 0; t < 4; ++t) {
      v16h b = load_bt_frag(BT, ldb, n0 + 16 * t, k0, lane);
      acc[t] = wmma_f32(a, b, acc[t]);
    }
  }
  int n = lane & 15, mg = (lane >> 4) * 8;
  if (blockIdx.y == 0) {
#pragma unroll
    for (int t = 0; t < 4; ++t)
#pragma unroll
      for (int v = 0; v < 8; ++v)
        Qd[(size_t)(m0 + mg + v) * 64 + 16 * t + n] = (_Float16)acc[t][v];
  } else {
    int bat = m0 >> 12;
    int rloc = (m0 & 4095) + mg;
#pragma unroll
    for (int t = 0; t < 4; ++t) {
      v8h p;
#pragma unroll
      for (int v = 0; v < 8; ++v) p[v] = (_Float16)acc[t][v];
      *(v8h*)(KTd + (size_t)bat * 64 * 4096 + (size_t)(16 * t + n) * 4096 + rloc) = p;
    }
  }
}

// ---------------- Fused: J = LayerNorm(Q @ M + J), maintains J and Jᵀ ---------
// Qb: 16384x64 f16. MT: per-batch 512x64 f16 (= Mᵀ, scale pre-applied).
// Block = 256 threads (8 waves), 16 rows x 512 cols per block.
__global__ void attn_update_ln_kernel(const _Float16* __restrict__ Qb,
                                      const _Float16* __restrict__ MT,
                                      _Float16* __restrict__ Jh,
                                      _Float16* __restrict__ JT,
                                      const float* __restrict__ gamma,
                                      const float* __restrict__ beta) {
  __shared__ float tile[16][512];
  int row0 = blockIdx.x * 16;
  int bat = row0 >> 12;
  int wave = threadIdx.x >> 5;
  int lane = threadIdx.x & 31;
  int n0 = wave * 64;
  const _Float16* Mb = MT + (size_t)bat * 512 * 64;
  v8f acc[4] = {};
#pragma unroll
  for (int k0 = 0; k0 < 64; k0 += 32) {
    v16h a = load_a_frag(Qb, 64, row0, k0, lane);
#pragma unroll
    for (int t = 0; t < 4; ++t) {
      v16h b = load_bt_frag(Mb, 64, n0 + 16 * t, k0, lane);
      acc[t] = wmma_f32(a, b, acc[t]);
    }
  }
  int n = lane & 15, mg = (lane >> 4) * 8;
#pragma unroll
  for (int t = 0; t < 4; ++t)
#pragma unroll
    for (int v = 0; v < 8; ++v) {
      int col = n0 + 16 * t + n;
      int r = mg + v;
      tile[r][col] = acc[t][v] + (float)Jh[(size_t)(row0 + r) * 512 + col];
    }
  __syncthreads();
  // LayerNorm: 16 threads per row (contiguous lanes within a wave), 32 cols each.
  int r = threadIdx.x >> 4;
  int s = threadIdx.x & 15;
  float sum = 0.f, sq = 0.f;
#pragma unroll
  for (int j = 0; j < 32; ++j) {
    float x = tile[r][s * 32 + j];
    sum += x; sq += x * x;
  }
#pragma unroll
  for (int m = 8; m >= 1; m >>= 1) {
    sum += __shfl_xor(sum, m, 32);
    sq  += __shfl_xor(sq,  m, 32);
  }
  float mu = sum * (1.0f / 512.0f);
  float var = sq * (1.0f / 512.0f) - mu * mu;
  float rstd = rsqrtf(var + 1e-5f);
#pragma unroll
  for (int j = 0; j < 32; ++j) {
    int col = s * 32 + j;
    float x = (tile[r][col] - mu) * rstd * gamma[col] + beta[col];
    Jh[(size_t)(row0 + r) * 512 + col] = (_Float16)x;
    tile[r][col] = x;  // stash normalized value for the transposed write
  }
  __syncthreads();
  // Jᵀ: each thread emits 2 columns as 16 contiguous halves (two 16B stores).
#pragma unroll
  for (int cc = 0; cc < 2; ++cc) {
    int col = threadIdx.x * 2 + cc;
    v8h p0, p1;
#pragma unroll
    for (int v = 0; v < 8; ++v) { p0[v] = (_Float16)tile[v][col]; p1[v] = (_Float16)tile[8 + v][col]; }
    _Float16* dst = JT + (size_t)bat * 512 * 4096 + (size_t)col * 4096 + (row0 & 4095);
    *(v8h*)dst = p0;
    *(v8h*)(dst + 8) = p1;
  }
}

// ---------------- small glue kernels -----------------------------------------

__global__ void pack_hf_kernel(const float* __restrict__ H, const float* __restrict__ F,
                               _Float16* __restrict__ HFh, _Float16* __restrict__ Fh,
                               _Float16* __restrict__ FT) {
  int i = blockIdx.x * blockDim.x + threadIdx.x;  // R*128
  int r = i >> 7, c = i & 127;
  float v = (c < 64) ? H[(size_t)r * 64 + c] : F[(size_t)r * 64 + (c - 64)];
  _Float16 hv = (_Float16)v;
  HFh[i] = hv;
  if (c >= 64) {
    Fh[(size_t)r * 64 + (c - 64)] = hv;
    int bat = r >> 12, nloc = r & 4095;
    FT[(size_t)bat * 64 * 4096 + (size_t)(c - 64) * 4096 + nloc] = hv;
  }
}

// dst (out_dim x in_dim, f16) = transpose of src (in_dim x out_dim, f32)
__global__ void transpose_w_kernel(const float* __restrict__ src, _Float16* __restrict__ dst,
                                   int in_dim, int out_dim) {
  int i = blockIdx.x * blockDim.x + threadIdx.x;
  if (i >= in_dim * out_dim) return;
  int o = i / in_dim, r = i - o * in_dim;
  dst[i] = (_Float16)src[(size_t)r * out_dim + o];
}

__global__ void make_g_kernel(const _Float16* __restrict__ Qh, const _Float16* __restrict__ Qf,
                              const float* __restrict__ wph, const float* __restrict__ wpf,
                              _Float16* __restrict__ G) {
  int i = blockIdx.x * blockDim.x + threadIdx.x;  // R*64
  G[i] = (_Float16)((float)Qh[i] * wph[0] + (float)Qf[i] * wpf[0]);
}

// RT[b][n][k] = scale * sum_j P[b][k][j] * Wout[j][n]   (R = scale*P@Wout, stored ᵀ)
__global__ void make_rt_kernel(const float* __restrict__ P, const float* __restrict__ Wout,
                               _Float16* __restrict__ RT, float scale) {
  int b = blockIdx.x;
  int n = threadIdx.x;  // 64
  for (int k = 0; k < 64; ++k) {
    float s = 0.f;
#pragma unroll
    for (int j = 0; j < 64; ++j) s += P[(size_t)b * 4096 + k * 64 + j] * Wout[(size_t)j * 64 + n];
    RT[(size_t)b * 4096 + n * 64 + k] = (_Float16)(s * scale);
  }
}

// ---------------- host orchestration -----------------------------------------

extern "C" void kernel_launch(void* const* d_in, const int* in_sizes, int n_in,
                              void* d_out, int out_size, void* d_ws, size_t ws_size,
                              hipStream_t stream) {
  const float* H       = (const float*)d_in[0];
  const float* F       = (const float*)d_in[1];
  const float* W_in    = (const float*)d_in[2];
  const float* b_in    = (const float*)d_in[3];
  const float* Wq_l    = (const float*)d_in[4];
  const float* Wk_l    = (const float*)d_in[5];
  const float* gamma_l = (const float*)d_in[6];
  const float* beta_l  = (const float*)d_in[7];
  const float* Wqh     = (const float*)d_in[8];
  const float* Wkh     = (const float*)d_in[9];
  const float* Wqf     = (const float*)d_in[10];
  const float* wph     = (const float*)d_in[11];
  const float* wpf     = (const float*)d_in[12];
  const float* W_out   = (const float*)d_in[13];
  const float* b_out   = (const float*)d_in[14];
  (void)in_sizes; (void)n_in; (void)out_size; (void)ws_size;

  constexpr int NB = 4, N = 4096, R = NB * N;  // 16384 total rows
  constexpr int L = 4;
  const float scale = 0.125f;  // 1/sqrt(d_k=64)

  char* w = (char*)d_ws;
  auto alloc = [&](size_t bytes) -> void* {
    void* p = (void*)w;
    w += (bytes + 255) & ~(size_t)255;
    return p;
  };
  _Float16* HFh   = (_Float16*)alloc((size_t)R * 128 * 2);
  _Float16* Fh    = (_Float16*)alloc((size_t)R * 64 * 2);
  _Float16* FT    = (_Float16*)alloc((size_t)NB * 64 * 4096 * 2);
  _Float16* Jh    = (_Float16*)alloc((size_t)R * 512 * 2);
  _Float16* JT    = (_Float16*)alloc((size_t)NB * 512 * 4096 * 2);
  _Float16* Qb    = (_Float16*)alloc((size_t)R * 64 * 2);
  _Float16* KT    = (_Float16*)alloc((size_t)NB * 64 * 4096 * 2);
  _Float16* Qf    = (_Float16*)alloc((size_t)R * 64 * 2);
  _Float16* G     = (_Float16*)alloc((size_t)R * 64 * 2);
  _Float16* MT    = (_Float16*)alloc((size_t)NB * 512 * 64 * 2);
  _Float16* WinT  = (_Float16*)alloc((size_t)512 * 128 * 2);
  _Float16* WqkT  = (_Float16*)alloc((size_t)L * 128 * 512 * 2);
  _Float16* WqkhT = (_Float16*)alloc((size_t)128 * 512 * 2);
  _Float16* WqfT  = (_Float16*)alloc((size_t)64 * 64 * 2);
  float*    Pb    = (float*)   alloc((size_t)NB * 64 * 64 * 4);
  _Float16* RT    = (_Float16*)alloc((size_t)NB * 64 * 64 * 2);

  // 1) pack activations to f16 (and Fᵀ), transpose weights to N x K f16
  pack_hf_kernel<<<dim3((R * 128) / 256), 256, 0, stream>>>(H, F, HFh, Fh, FT);
  transpose_w_kernel<<<(512 * 128 + 255) / 256, 256, 0, stream>>>(W_in, WinT, 128, 512);
  for (int t = 0; t < L; ++t) {
    transpose_w_kernel<<<(512 * 64 + 255) / 256, 256, 0, stream>>>(
        Wq_l + (size_t)t * 512 * 64, WqkT + (size_t)t * 128 * 512, 512, 64);
    transpose_w_kernel<<<(512 * 64 + 255) / 256, 256, 0, stream>>>(
        Wk_l + (size_t)t * 512 * 64, WqkT + (size_t)t * 128 * 512 + 64 * 512, 512, 64);
  }
  transpose_w_kernel<<<(512 * 64 + 255) / 256, 256, 0, stream>>>(Wqh, WqkhT, 512, 64);
  transpose_w_kernel<<<(512 * 64 + 255) / 256, 256, 0, stream>>>(Wkh, WqkhT + 64 * 512, 512, 64);
  transpose_w_kernel<<<(64 * 64 + 255) / 256, 256, 0, stream>>>(Wqf, WqfT, 64, 64);

  // 2) J = [H|F] @ W_in + b_in     (also writes Jᵀ via transposed store path)
  gemm_bt_kernel<true><<<dim3(R / 16, 512 / 64, 1), 32, 0, stream>>>(
      HFh, 128, 0, WinT, 128, 0, Jh, 512, 0, b_in, 1.0f, 128,
      JT, (size_t)512 * 4096);

  // 3) layers: J <- LN( Q (KᵀJ) scale + J )
  for (int t = 0; t < L; ++t) {
    // Q = J@Wq (row-major), K = J@Wk stored directly as Kᵀ (B x 64 x 4096)
    gemm_qkt_kernel<<<dim3(R / 16, 2, 1), 32, 0, stream>>>(
        Jh, 512, WqkT + (size_t)t * 128 * 512, 512, Qb, KT, 512);
    // Mᵀ[d][k] = scale * sum_n Jᵀ[d][n] Kᵀ[k][n]  — all-contiguous gemm_bt
    gemm_bt_kernel<true><<<dim3(512 / 16, 1, NB), 32, 0, stream>>>(
        JT, 4096, (size_t)512 * 4096, KT, 4096, (size_t)64 * 4096,
        MT, 64, (size_t)512 * 64, nullptr, scale, 4096, nullptr, 0);
    // J = LN(Q @ M + J); refreshes Jᵀ
    attn_update_ln_kernel<<<dim3(R / 16), 256, 0, stream>>>(
        Qb, MT, Jh, JT, gamma_l + t * 512, beta_l + t * 512);
  }

  // 4) readout: Q_h (row-major) | K_h (stored as K_hᵀ),  Q_f = F @ Wqf
  gemm_qkt_kernel<<<dim3(R / 16, 2, 1), 32, 0, stream>>>(
      Jh, 512, WqkhT, 512, Qb, KT, 512);
  gemm_bt_kernel<true><<<dim3(R / 16, 1, 1), 32, 0, stream>>>(
      Fh, 64, 0, WqfT, 64, 0, Qf, 64, 0, nullptr, 1.0f, 64, nullptr, 0);
  // G = wph*Q_h + wpf*Q_f
  make_g_kernel<<<(R * 64) / 256, 256, 0, stream>>>(Qb, Qf, wph, wpf, G);
  // P[k][j] = sum_n K_hᵀ[k][n] Fᵀ[j][n]  (per batch 64x64) — contiguous gemm_bt
  gemm_bt_kernel<false><<<dim3(64 / 16, 1, NB), 32, 0, stream>>>(
      KT, 4096, (size_t)64 * 4096, FT, 4096, (size_t)64 * 4096,
      Pb, 64, (size_t)64 * 64, nullptr, 1.0f, 4096, nullptr, 0);
  // Rᵀ = (scale * P @ W_out)ᵀ
  make_rt_kernel<<<NB, 64, 0, stream>>>(Pb, W_out, RT, scale);
  // out = G @ R + b_out  (per-batch R, f32 output)
  gemm_bt_kernel<false><<<dim3(N / 16, 1, NB), 32, 0, stream>>>(
      G, 64, (size_t)N * 64, RT, 64, (size_t)64 * 64,
      (float*)d_out, 64, (size_t)N * 64, b_out, 1.0f, 64, nullptr, 0);
}
